// MusicTransformer_61778809586264
// MI455X (gfx1250) — matrix operand order repaired and checked by reading, not compiled, not run on hardware
//
#include <hip/hip_runtime.h>
#include <hip/hip_bf16.h>
#include <math.h>

// ---------------------------------------------------------------------------
// MI455X (gfx1250) Music Transformer forward.
// All GEMM-shaped work via v_wmma_f32_16x16x32_bf16 (fp32 accumulate,
// native-cvt fp32->bf16 staging). Compute-bound (~413 GFLOP); model weights
// (~76MB) live in the 192MB L2, so matrix-core throughput decides.
// ---------------------------------------------------------------------------

typedef __attribute__((ext_vector_type(16))) __bf16         bf16x16;
typedef __attribute__((ext_vector_type(8)))  float          f32x8;
typedef __attribute__((ext_vector_type(8)))  unsigned short u16x8;

union FragBF {
    bf16x16        v;
    u16x8          h8[2];
    unsigned short u[16];
};

// native hardware conversion (v_cvt_*bf16_f32 class on gfx1250)
__device__ inline unsigned short f2bf(float f) {
    union { __bf16 b; unsigned short u; } x;
    x.b = (__bf16)f;
    return x.u;
}

// load 8 consecutive fp32 -> 8 bf16 (two float4 vector loads + native cvt)
__device__ inline void ld8bf(const float* __restrict__ p, unsigned short* dst) {
    float4 f0 = ((const float4*)p)[0];
    float4 f1 = ((const float4*)p)[1];
    dst[0] = f2bf(f0.x); dst[1] = f2bf(f0.y); dst[2] = f2bf(f0.z); dst[3] = f2bf(f0.w);
    dst[4] = f2bf(f1.x); dst[5] = f2bf(f1.y); dst[6] = f2bf(f1.z); dst[7] = f2bf(f1.w);
}

__device__ inline f32x8 wmma_bf16(const FragBF& a, const FragBF& b, f32x8 c) {
    return __builtin_amdgcn_wmma_f32_16x16x32_bf16(
        /*neg_a=*/false, a.v, /*neg_b=*/false, b.v,
        /*c_mod=*/(short)0, c, /*reuse_a=*/false, /*reuse_b=*/false);
}

// ---------------------------------------------------------------------------
// GEMM:  C[M,N] = act(A[M,K] @ W[K,N] + bias)    (all fp32 in memory)
// Block: 256 threads = 8 waves; block tile 64x128; wave tile 32x32 (2x2 WMMA).
// ---------------------------------------------------------------------------
__global__ __launch_bounds__(256) void gemm_bias_kernel(
    const float* __restrict__ A, const float* __restrict__ W,
    const float* __restrict__ bias, float* __restrict__ C,
    int M, int N, int K, int act)
{
    __shared__ unsigned short As[64 * 32];    // [m][k] bf16
    __shared__ unsigned short Bs[128 * 32];   // [n][k] bf16 (transposed)

    const int tid  = threadIdx.x;
    const int lane = tid & 31;
    const int wid  = tid >> 5;
    const int half = lane >> 4;
    const int l16  = lane & 15;

    const int bm = (int)blockIdx.y * 64;
    const int bn = (int)blockIdx.x * 128;
    const int waveM = (wid & 1) * 32;
    const int waveN = (wid >> 1) * 32;

    // staging coordinates
    const int ar = tid >> 2;            // A row 0..63
    const int ac = (tid & 3) * 8;       // A col base {0,8,16,24}
    const int br = tid >> 3;            // B (k) row 0..31
    const int bc = (tid & 7) * 16;      // B (n) col base

    f32x8 acc[2][2];
    for (int i = 0; i < 2; ++i)
        for (int j = 0; j < 2; ++j)
            for (int e = 0; e < 8; ++e) acc[i][j][e] = 0.0f;

    for (int k0 = 0; k0 < K; k0 += 32) {
        // stage A tile 64x32: two b128 loads + native cvt + one ds_store_b128
        {
            const float* src = &A[(size_t)(bm + ar) * K + (k0 + ac)];
            float4 f0 = ((const float4*)src)[0];
            float4 f1 = ((const float4*)src)[1];
            u16x8 pk;
            pk[0] = f2bf(f0.x); pk[1] = f2bf(f0.y); pk[2] = f2bf(f0.z); pk[3] = f2bf(f0.w);
            pk[4] = f2bf(f1.x); pk[5] = f2bf(f1.y); pk[6] = f2bf(f1.z); pk[7] = f2bf(f1.w);
            *(u16x8*)&As[ar * 32 + ac] = pk;
        }
        // stage B tile 32x128 transposed -> Bs[n][k]
        {
            const float* src = &W[(size_t)(k0 + br) * N + (bn + bc)];
            for (int t = 0; t < 4; ++t) {
                float4 f = ((const float4*)src)[t];
                Bs[(bc + t * 4 + 0) * 32 + br] = f2bf(f.x);
                Bs[(bc + t * 4 + 1) * 32 + br] = f2bf(f.y);
                Bs[(bc + t * 4 + 2) * 32 + br] = f2bf(f.z);
                Bs[(bc + t * 4 + 3) * 32 + br] = f2bf(f.w);
            }
        }
        // warm L2/L0 for the next K tile while WMMAs run (global_prefetch_b8)
        if (k0 + 32 < K) {
            __builtin_prefetch(&A[(size_t)(bm + ar) * K + (k0 + 32 + ac)], 0, 1);
            __builtin_prefetch(&W[(size_t)(k0 + 32 + br) * N + (bn + bc)], 0, 1);
        }
        __syncthreads();

        FragBF a[2], b[2];
        for (int t = 0; t < 2; ++t) {
            // A frag: k runs of 8 at {half*8, 16+half*8}
            int row = waveM + t * 16 + l16;
            a[t].h8[0] = *(const u16x8*)&As[row * 32 + half * 8];
            a[t].h8[1] = *(const u16x8*)&As[row * 32 + 16 + half * 8];
            // B frag: k runs of 8 at {half*8, 16+half*8}
            int col = waveN + t * 16 + l16;
            b[t].h8[0] = *(const u16x8*)&Bs[col * 32 + half * 8];
            b[t].h8[1] = *(const u16x8*)&Bs[col * 32 + 16 + half * 8];
        }
        for (int i = 0; i < 2; ++i)
            for (int j = 0; j < 2; ++j)
                acc[i][j] = wmma_bf16(a[i], b[j], acc[i][j]);
        __syncthreads();
    }

    for (int i = 0; i < 2; ++i)
        for (int j = 0; j < 2; ++j) {
            int n  = bn + waveN + j * 16 + l16;
            float bb = bias[n];
            for (int e = 0; e < 8; ++e) {
                int m = bm + waveM + i * 16 + e + half * 8;
                float v = acc[i][j][e] + bb;
                if (act) v = v > 0.0f ? v : 0.0f;
                C[(size_t)m * N + n] = v;
            }
        }
}

// ---------------------------------------------------------------------------
// Flash-style attention with Music-Transformer relative positions.
// One wave handles a 16-row query tile, streaming j in steps of 32.
// Q/K/V/O layout: [B,S,D] with head h slice at column offset h*DH.
// ---------------------------------------------------------------------------
__global__ __launch_bounds__(32) void attn_kernel(
    const float* __restrict__ Q, const float* __restrict__ K,
    const float* __restrict__ V, const float* __restrict__ PE,
    float* __restrict__ O, int S, int Dm, int DH)
{
    __shared__ float          sc[16][32];     // combined scores tile
    __shared__ float          Rt[16][48];     // rel-position band q@relk^T
    __shared__ unsigned short P[16][32];      // probabilities (bf16)
    __shared__ unsigned short Vt[64][32];     // V tile transposed [d][j] bf16
    __shared__ float          rowm[16], rowl[16], rowcorr[16];

    const int lane = threadIdx.x;
    const int half = lane >> 4;
    const int l16  = lane & 15;
    const int i0 = (int)blockIdx.x * 16;
    const int h  = (int)blockIdx.y;
    const int b  = (int)blockIdx.z;
    const float inv_sqrt_dh = rsqrtf((float)DH);

    const float* qb  = Q  + (size_t)b * S * Dm + h * DH;
    const float* kb  = K  + (size_t)b * S * Dm + h * DH;
    const float* vb  = V  + (size_t)b * S * Dm + h * DH;
    const float* peb = PE + h * DH;            // relk[h][r][d] = PE[r][h*DH+d]

    // Q fragments (A operand), kept in registers for all j
    FragBF qa[2];
    {
        const float* qr = qb + (size_t)(i0 + l16) * Dm;
        for (int kc = 0; kc < 2; ++kc) {
            ld8bf(qr + kc * 32 + half * 8,      &qa[kc].u[0]);
            ld8bf(qr + kc * 32 + 16 + half * 8, &qa[kc].u[8]);
        }
    }

    if (lane < 16) { rowm[lane] = -1e30f; rowl[lane] = 0.0f; }

    f32x8 oacc[4];
    for (int nc = 0; nc < 4; ++nc)
        for (int e = 0; e < 8; ++e) oacc[nc][e] = 0.0f;

    for (int j0 = 0; j0 < S; j0 += 32) {
        // ---- scores = q @ k^T (16x32) ----
        f32x8 s_[2];
        for (int nc = 0; nc < 2; ++nc) {
            for (int e = 0; e < 8; ++e) s_[nc][e] = 0.0f;
            const float* kr = kb + (size_t)(j0 + nc * 16 + l16) * Dm;
            for (int kc = 0; kc < 2; ++kc) {
                FragBF bk_;
                ld8bf(kr + kc * 32 + half * 8,      &bk_.u[0]);
                ld8bf(kr + kc * 32 + 16 + half * 8, &bk_.u[8]);
                s_[nc] = wmma_bf16(qa[kc], bk_, s_[nc]);
            }
        }

        // ---- rel band: Rt = q @ relk[rbase..rbase+47]^T ----
        const int rbase = i0 - j0 + S - 32;
        for (int nc = 0; nc < 3; ++nc) {
            f32x8 racc;
            for (int e = 0; e < 8; ++e) racc[e] = 0.0f;
            int r = rbase + nc * 16 + l16;
            r = r < 0 ? 0 : (r > S - 1 ? S - 1 : r);   // clamped rows feed only masked entries
            const float* rr = peb + (size_t)r * Dm;
            for (int kc = 0; kc < 2; ++kc) {
                FragBF rb;
                ld8bf(rr + kc * 32 + half * 8,      &rb.u[0]);
                ld8bf(rr + kc * 32 + 16 + half * 8, &rb.u[8]);
                racc = wmma_bf16(qa[kc], rb, racc);
            }
            for (int e = 0; e < 8; ++e)
                Rt[e + half * 8][nc * 16 + l16] = racc[e];
        }

        // ---- stage V tile transposed: Vt[d][j'] ----
        {
            const float* vr = vb + (size_t)(j0 + lane) * Dm;
            for (int d4 = 0; d4 < 16; ++d4) {
                float4 f = ((const float4*)vr)[d4];
                Vt[d4 * 4 + 0][lane] = f2bf(f.x);
                Vt[d4 * 4 + 1][lane] = f2bf(f.y);
                Vt[d4 * 4 + 2][lane] = f2bf(f.z);
                Vt[d4 * 4 + 3][lane] = f2bf(f.w);
            }
        }
        __syncthreads();

        // ---- combine scores + gathered rel, scale ----
        for (int nc = 0; nc < 2; ++nc)
            for (int e = 0; e < 8; ++e) {
                int m  = e + half * 8;
                int jp = nc * 16 + l16;
                int jg = j0 + jp, ig = i0 + m;
                float rel = (jg >= ig) ? Rt[m][m - jp + 31] : 0.0f;
                sc[m][jp] = (s_[nc][e] + rel) * inv_sqrt_dh;
            }
        __syncthreads();

        // ---- online softmax (lanes 0..15 own one row each) ----
        if (lane < 16) {
            int m = lane;
            float mo = rowm[m];
            float mx = mo;
            for (int jp = 0; jp < 32; ++jp) mx = fmaxf(mx, sc[m][jp]);
            float corr = __expf(mo - mx);
            float lsum = rowl[m] * corr;
            for (int jp = 0; jp < 32; ++jp) {
                float p = __expf(sc[m][jp] - mx);
                P[m][jp] = f2bf(p);
                lsum += p;
            }
            rowm[m] = mx; rowl[m] = lsum; rowcorr[m] = corr;
        }
        __syncthreads();

        // ---- rescale O, then O += P @ V ----
        float cf[8];
        for (int e = 0; e < 8; ++e) cf[e] = rowcorr[e + half * 8];
        for (int nc = 0; nc < 4; ++nc)
            for (int e = 0; e < 8; ++e) oacc[nc][e] *= cf[e];

        FragBF pa;
        pa.h8[0] = *(const u16x8*)&P[l16][half * 8];
        pa.h8[1] = *(const u16x8*)&P[l16][16 + half * 8];
        for (int nc = 0; nc < 4; ++nc) {
            FragBF vf;
            int n = nc * 16 + l16;               // d column
            vf.h8[0] = *(const u16x8*)&Vt[n][half * 8];
            vf.h8[1] = *(const u16x8*)&Vt[n][16 + half * 8];
            oacc[nc] = wmma_bf16(pa, vf, oacc[nc]);
        }
        __syncthreads();
    }

    // ---- finalize: divide by row sums, write O[b, i, h*DH + d] ----
    float linv[8];
    for (int e = 0; e < 8; ++e) linv[e] = 1.0f / rowl[e + half * 8];
    for (int nc = 0; nc < 4; ++nc)
        for (int e = 0; e < 8; ++e) {
            int m = i0 + e + half * 8;
            int d = nc * 16 + l16;
            O[(size_t)(b * S + m) * Dm + h * DH + d] = oacc[nc][e] * linv[e];
        }
}

// ---------------------------------------------------------------------------
// Residual + LayerNorm: Y = LN(X + R) * g + be   (D = 512, 256 thr x 2 elems)
// ---------------------------------------------------------------------------
__global__ __launch_bounds__(256) void ln_kernel(
    const float* __restrict__ X, const float* __restrict__ R,
    const float* __restrict__ g, const float* __restrict__ be,
    float* __restrict__ Y)
{
    __shared__ float buf[256];
    const int row = blockIdx.x;
    const int tid = threadIdx.x;
    const float* x = X + (size_t)row * 512;
    const float* r = R + (size_t)row * 512;

    float v0 = x[tid] + r[tid];
    float v1 = x[tid + 256] + r[tid + 256];

    buf[tid] = v0 + v1;
    __syncthreads();
    for (int st = 128; st > 0; st >>= 1) {
        if (tid < st) buf[tid] += buf[tid + st];
        __syncthreads();
    }
    float mean = buf[0] * (1.0f / 512.0f);
    __syncthreads();

    float d0 = v0 - mean, d1 = v1 - mean;
    buf[tid] = d0 * d0 + d1 * d1;
    __syncthreads();
    for (int st = 128; st > 0; st >>= 1) {
        if (tid < st) buf[tid] += buf[tid + st];
        __syncthreads();
    }
    float rstd = rsqrtf(buf[0] * (1.0f / 512.0f) + 1e-5f);

    float* y = Y + (size_t)row * 512;
    y[tid]       = d0 * rstd * g[tid] + be[tid];
    y[tid + 256] = d1 * rstd * g[tid + 256] + be[tid + 256];
}

// ---------------------------------------------------------------------------
// Sinusoidal positional-encoding buffer: PE[r, 2k]=sin(r*div_k), PE[r,2k+1]=cos
// ---------------------------------------------------------------------------
__global__ __launch_bounds__(256) void pe_kernel(float* __restrict__ pe, int S, int Dm)
{
    int idx = blockIdx.x * 256 + threadIdx.x;
    int half = Dm >> 1;
    if (idx >= S * half) return;
    int r = idx / half, kp = idx % half;
    float div = __expf(-logf(10000.0f) * (2.0f * (float)kp) / (float)Dm);
    float ang = (float)r * div;
    pe[(size_t)r * Dm + 2 * kp]     = __sinf(ang);
    pe[(size_t)r * Dm + 2 * kp + 1] = __cosf(ang);
}

// ---------------------------------------------------------------------------
extern "C" void kernel_launch(void* const* d_in, const int* in_sizes, int n_in,
                              void* d_out, int out_size, void* d_ws, size_t ws_size,
                              hipStream_t stream)
{
    (void)in_sizes; (void)n_in; (void)out_size; (void)ws_size;

    const float* x     = (const float*)d_in[0];
    const float* emb_W = (const float*)d_in[1];
    const float* emb_b = (const float*)d_in[2];
    const float* Wq    = (const float*)d_in[3];
    const float* bq    = (const float*)d_in[4];
    const float* Wk    = (const float*)d_in[5];
    const float* bk    = (const float*)d_in[6];
    const float* Wv    = (const float*)d_in[7];
    const float* bv    = (const float*)d_in[8];
    const float* Wo    = (const float*)d_in[9];
    const float* bo    = (const float*)d_in[10];
    const float* W1    = (const float*)d_in[11];
    const float* b1    = (const float*)d_in[12];
    const float* W2    = (const float*)d_in[13];
    const float* b2    = (const float*)d_in[14];
    const float* g1    = (const float*)d_in[15];
    const float* be1   = (const float*)d_in[16];
    const float* g2    = (const float*)d_in[17];
    const float* be2   = (const float*)d_in[18];
    const float* oW    = (const float*)d_in[19];
    const float* ob    = (const float*)d_in[20];
    float* out = (float*)d_out;

    constexpr int Bz = 8, S = 1024, IN = 128, OUT = 128;
    constexpr int D = 512, H = 8, L = 6, FF = 2048, DH = 64;
    constexpr int M = Bz * S;   // 8192

    float* ws = (float*)d_ws;
    float* hA = ws;
    float* hB = hA + (size_t)M * D;
    float* q  = hB + (size_t)M * D;
    float* kk = q  + (size_t)M * D;
    float* vv = kk + (size_t)M * D;
    float* ao = vv + (size_t)M * D;
    float* ff = ao + (size_t)M * D;
    float* pe = ff + (size_t)M * FF;

    pe_kernel<<<dim3((S * (D / 2) + 255) / 256), dim3(256), 0, stream>>>(pe, S, D);

    auto gemm = [&](const float* Ain, const float* Win, const float* bin,
                    float* Cout, int m, int n, int kdim, int act) {
        gemm_bias_kernel<<<dim3(n / 128, m / 64), dim3(256), 0, stream>>>(
            Ain, Win, bin, Cout, m, n, kdim, act);
    };

    // embedding
    gemm(x, emb_W, emb_b, hA, M, D, IN, 0);

    for (int l = 0; l < L; ++l) {
        gemm(hA, Wq + (size_t)l * D * D, bq + (size_t)l * D, q,  M, D, D, 0);
        gemm(hA, Wk + (size_t)l * D * D, bk + (size_t)l * D, kk, M, D, D, 0);
        gemm(hA, Wv + (size_t)l * D * D, bv + (size_t)l * D, vv, M, D, D, 0);

        attn_kernel<<<dim3(S / 16, H, Bz), dim3(32), 0, stream>>>(
            q, kk, vv, pe, ao, S, D, DH);

        gemm(ao, Wo + (size_t)l * D * D, bo + (size_t)l * D, q, M, D, D, 0);   // q is dead, reuse
        ln_kernel<<<dim3(M), dim3(256), 0, stream>>>(hA, q, g1 + (size_t)l * D, be1 + (size_t)l * D, hB);

        gemm(hB, W1 + (size_t)l * D * FF, b1 + (size_t)l * FF, ff, M, FF, D, 1);
        gemm(ff, W2 + (size_t)l * FF * D, b2 + (size_t)l * D, kk, M, D, FF, 0); // kk dead, reuse
        ln_kernel<<<dim3(M), dim3(256), 0, stream>>>(hB, kk, g2 + (size_t)l * D, be2 + (size_t)l * D, hA);
    }

    // output projection
    gemm(hA, oW, ob, out, M, OUT, D, 0);
}
